// PointTransformerSegmentation_78013785964579
// MI455X (gfx1250) — matrix operand order repaired and checked by reading, not compiled, not run on hardware
//
#include <hip/hip_runtime.h>
#include <hip/hip_bf16.h>
#include <cstddef>

typedef __attribute__((ext_vector_type(16))) _Float16 v16h;
typedef __attribute__((ext_vector_type(4)))  _Float16 v4h;
typedef __attribute__((ext_vector_type(2)))  _Float16 v2h;
typedef __attribute__((ext_vector_type(8)))  float    v8f;
typedef __attribute__((ext_vector_type(4)))  float    v4f;

#define BB 2          // batch
#define NPTS0 4096    // points at full resolution

// =====================================================================
// FAST GEMM: requires M%16==0, K%32==0, N%16==0  (true for all d->d
// layers, 32..512). One wave per 16x16 tile; no bounds checks, vector
// loads, LDS-staged f16 tiles, f32 accumulate in the WMMA C/D chain.
//   A tile As[row][k]   : rows padded to 36 f16 (18 dwords; 16 reading
//   B tile BsT[col][k]  :  lanes hit 16 distinct banks), v4h stores,
//                          v2h fragment reads (contiguous K pairs).
// =====================================================================
template<bool RELU, bool RES>
__global__ __launch_bounds__(32)
void gemm_wmma_fast(const float* __restrict__ X, const float* __restrict__ W,
                    const float* __restrict__ bias, const float* __restrict__ res,
                    float* __restrict__ Y, int M, int K, int N)
{
  __shared__ __align__(16) _Float16 As [16][36];
  __shared__ __align__(16) _Float16 BsT[16][36];
  const int lane = threadIdx.x;
  const int r = lane & 15;
  const int h = lane >> 4;
  const int row0 = blockIdx.x * 16;
  const int col0 = blockIdx.y * 16;
  const int col  = col0 + r;

  float bv = bias[col];                 // bias folded into accumulator once
  v8f acc;
  #pragma unroll
  for (int j = 0; j < 8; ++j) acc[j] = bv;

  // per-lane load coordinates (constant over k-chunks)
  const int ar = lane >> 3;             // + it*4  -> A row
  const int ak = (lane & 7) * 4;        //          A k (float4)
  const int kr = lane >> 2;             // + it*8  -> B k row
  const int nc = (lane & 3) * 4;        //          B col (float4)

  for (int k0 = 0; k0 < K; k0 += 32) {
    #pragma unroll
    for (int it = 0; it < 4; ++it) {
      // A: 16x32 tile, 128B coalesced float4 per lane, 16B aligned (K%32==0)
      int arow = it * 4 + ar;
      v4f xa = *(const v4f*)&X[(size_t)(row0 + arow) * K + k0 + ak];
      v4h ha;
      ha[0] = (_Float16)xa[0]; ha[1] = (_Float16)xa[1];
      ha[2] = (_Float16)xa[2]; ha[3] = (_Float16)xa[3];
      *(v4h*)&As[arow][ak] = ha;
      // B: 32x16 tile, float4 over 4 consecutive cols, scatter-transpose
      int krow = it * 8 + kr;
      v4f wb = *(const v4f*)&W[(size_t)(k0 + krow) * N + col0 + nc];
      BsT[nc + 0][krow] = (_Float16)wb[0];
      BsT[nc + 1][krow] = (_Float16)wb[1];
      BsT[nc + 2][krow] = (_Float16)wb[2];
      BsT[nc + 3][krow] = (_Float16)wb[3];
    }
    __syncthreads();

    // fragments per CDNA5 wave32 layout (ISA 7.12.2); all reads are v2h
    v16h a, bm;
    #pragma unroll
    for (int j = 0; j < 8; ++j) {
      int ka = (j < 4 ? 2 * j : 16 + 2 * (j - 4)) + 8 * h;  // A K pair
      v2h pa = *(const v2h*)&As[r][ka];
      a[2 * j] = pa[0]; a[2 * j + 1] = pa[1];
      int kb = (j < 4 ? 2 * j : 16 + 2 * (j - 4)) + 8 * h;  // B K pair
      v2h pb = *(const v2h*)&BsT[r][kb];
      bm[2 * j] = pb[0]; bm[2 * j + 1] = pb[1];
    }
    acc = __builtin_amdgcn_wmma_f32_16x16x32_f16(false, a, false, bm,
                                                 (short)0, acc, false, false);
    __syncthreads();
  }

  #pragma unroll
  for (int j = 0; j < 8; ++j) {
    int row = row0 + j + 8 * h;
    float v = acc[j];
    if (RES)  v += res[(size_t)row * N + col];
    if (RELU) v = fmaxf(v, 0.0f);
    Y[(size_t)row * N + col] = v;
  }
}

// =====================================================================
// Generic guarded GEMM for the ragged layers only (K=3, K=6, N=13).
// =====================================================================
template<bool RELU, bool RES>
__global__ __launch_bounds__(32)
void gemm_wmma_edge(const float* __restrict__ X, const float* __restrict__ W,
                    const float* __restrict__ bias, const float* __restrict__ res,
                    float* __restrict__ Y, int M, int K, int N)
{
  __shared__ _Float16 As[16][33];
  __shared__ _Float16 Bs[32][17];
  const int lane = threadIdx.x;
  const int r = lane & 15, h = lane >> 4;
  const int row0 = blockIdx.x * 16;
  const int col0 = blockIdx.y * 16;
  const int col  = col0 + r;

  float bv = (col < N) ? bias[col] : 0.0f;
  v8f acc;
  #pragma unroll
  for (int j = 0; j < 8; ++j) acc[j] = bv;

  for (int k0 = 0; k0 < K; k0 += 32) {
    #pragma unroll 4
    for (int t = 0; t < 16; ++t) {
      int gr = row0 + t, gk = k0 + lane;
      As[t][lane] = (gr < M && gk < K) ? (_Float16)X[(size_t)gr * K + gk]
                                       : (_Float16)0.0f;
    }
    #pragma unroll 4
    for (int t = 0; t < 16; ++t) {
      int i = t * 32 + lane;
      int br = i >> 4, bc = i & 15;
      int gk = k0 + br, gn = col0 + bc;
      Bs[br][bc] = (gk < K && gn < N) ? (_Float16)W[(size_t)gk * N + gn]
                                      : (_Float16)0.0f;
    }
    __syncthreads();
    v16h a, bm;
    #pragma unroll
    for (int j = 0; j < 8; ++j) {
      int ka = (j < 4 ? 2 * j : 16 + 2 * (j - 4)) + 8 * h;
      a[2 * j]  = As[r][ka];  a[2 * j + 1]  = As[r][ka + 1];
      int kb = (j < 4 ? 2 * j : 16 + 2 * (j - 4)) + 8 * h;
      bm[2 * j] = Bs[kb][r];  bm[2 * j + 1] = Bs[kb + 1][r];
    }
    acc = __builtin_amdgcn_wmma_f32_16x16x32_f16(false, a, false, bm,
                                                 (short)0, acc, false, false);
    __syncthreads();
  }

  #pragma unroll
  for (int j = 0; j < 8; ++j) {
    int row = row0 + j + 8 * h;
    if (row < M && col < N) {
      float v = acc[j];
      if (RES)  v += res[(size_t)row * N + col];
      if (RELU) v = fmaxf(v, 0.0f);
      Y[(size_t)row * N + col] = v;
    }
  }
}

// =====================================================================
// kNN with LDS-tiled reference points (512 pts = 6KB per chunk).
// Per-thread register top-K insertion (K<=16), stable on ties.
// Idle threads (n>=Nq) stay alive for cooperative loads + barriers.
// =====================================================================
#define KNN_TILE 512
template<int KN, bool WD>
__global__ __launch_bounds__(128)
void knn_kernel(const float* __restrict__ qxyz,
                const float* __restrict__ rxyz,
                int Nq, int Nr,
                int* __restrict__ idx, float* __restrict__ dst)
{
  __shared__ float rxs[KNN_TILE * 3];
  const int b = blockIdx.y;
  const int n = blockIdx.x * blockDim.x + threadIdx.x;
  const bool act = (n < Nq);
  const int nq = act ? n : 0;
  const float* q = qxyz + ((size_t)b * Nq + nq) * 3;
  const float qx = q[0], qy = q[1], qz = q[2];
  const float* ref = rxyz + (size_t)b * Nr * 3;

  float bd[KN]; int bi[KN];
  #pragma unroll
  for (int i = 0; i < KN; ++i) { bd[i] = 3.0e38f; bi[i] = 0; }

  for (int c0 = 0; c0 < Nr; c0 += KNN_TILE) {
    int cnt = Nr - c0; if (cnt > KNN_TILE) cnt = KNN_TILE;
    for (int i = threadIdx.x; i < cnt * 3; i += blockDim.x)
      rxs[i] = ref[(size_t)c0 * 3 + i];
    __syncthreads();
    for (int j = 0; j < cnt; ++j) {
      float dx = qx - rxs[j * 3 + 0];
      float dy = qy - rxs[j * 3 + 1];
      float dz = qz - rxs[j * 3 + 2];
      float d = dx * dx + dy * dy + dz * dz;
      if (d < bd[KN - 1]) {
        int p = KN - 1;
        #pragma unroll
        for (int s = KN - 1; s > 0; --s) {
          if (bd[s - 1] > d) { bd[s] = bd[s - 1]; bi[s] = bi[s - 1]; p = s - 1; }
          else break;
        }
        bd[p] = d; bi[p] = c0 + j;
      }
    }
    __syncthreads();
  }
  if (act) {
    int* oi = idx + ((size_t)b * Nq + n) * KN;
    #pragma unroll
    for (int i = 0; i < KN; ++i) oi[i] = bi[i];
    if (WD) {
      float* od = dst + ((size_t)b * Nq + n) * KN;
      #pragma unroll
      for (int i = 0; i < KN; ++i) od[i] = bd[i];
    }
  }
}

// =====================================================================
// Farthest point sampling: one block per batch, min-distance array in
// LDS (16KB of the 320KB WGP pool), argmax via shared reduction.
// =====================================================================
__global__ __launch_bounds__(256)
void fps_kernel(const float* __restrict__ xyz, int N, int m, int* __restrict__ out)
{
  __shared__ float dist[NPTS0];
  __shared__ float smax[256];
  __shared__ int   sarg[256];
  __shared__ int   sfar;
  const int b = blockIdx.x, tid = threadIdx.x, T = blockDim.x;
  const float* x = xyz + (size_t)b * N * 3;
  for (int i = tid; i < N; i += T) dist[i] = 1.0e10f;
  if (tid == 0) sfar = 0;
  __syncthreads();
  for (int it = 0; it < m; ++it) {
    int far = sfar;
    if (tid == 0) out[b * m + it] = far;
    float fx = x[far * 3], fy = x[far * 3 + 1], fz = x[far * 3 + 2];
    float mx = -1.0f; int ma = 0;
    for (int i = tid; i < N; i += T) {
      float dx = x[i * 3] - fx, dy = x[i * 3 + 1] - fy, dz = x[i * 3 + 2] - fz;
      float d = dx * dx + dy * dy + dz * dz;
      float nd = fminf(dist[i], d);
      dist[i] = nd;
      if (nd > mx) { mx = nd; ma = i; }
    }
    smax[tid] = mx; sarg[tid] = ma;
    __syncthreads();
    for (int s = T >> 1; s > 0; s >>= 1) {
      if (tid < s) {
        bool take = (smax[tid + s] > smax[tid]) ||
                    (smax[tid + s] == smax[tid] && sarg[tid + s] < sarg[tid]);
        if (take) { smax[tid] = smax[tid + s]; sarg[tid] = sarg[tid + s]; }
      }
      __syncthreads();
    }
    if (tid == 0) sfar = sarg[0];
    __syncthreads();
  }
}

// ------------------ elementwise / gather helpers ------------------
__global__ void extract_xyz_kernel(const float* __restrict__ pts,
                                   float* __restrict__ xyz, int N)
{
  int b = blockIdx.y;
  int e = blockIdx.x * blockDim.x + threadIdx.x;
  if (e >= N * 3) return;
  int n = e / 3, c = e % 3;
  xyz[((size_t)b * N + n) * 3 + c] = pts[((size_t)b * N + n) * 6 + c];
}

__global__ void gather_feat_kernel(const float* __restrict__ f,
                                   const int* __restrict__ idx,
                                   float* __restrict__ out,
                                   int Nq, int Nr, int K, int C)
{
  int b = blockIdx.y;
  long long e = (long long)blockIdx.x * blockDim.x + threadIdx.x;
  long long total = (long long)Nq * K * C;
  if (e >= total) return;
  int c = (int)(e % C);
  long long nk = e / C;
  int k = (int)(nk % K);
  int n = (int)(nk / K);
  int j = idx[((size_t)b * Nq + n) * K + k];
  out[(((size_t)b * Nq + n) * K + k) * C + c] = f[((size_t)b * Nr + j) * C + c];
}

__global__ void gather_xyz_kernel(const float* __restrict__ xyz,
                                  const int* __restrict__ fidx,
                                  float* __restrict__ out, int N, int m)
{
  int b = blockIdx.y;
  int e = blockIdx.x * blockDim.x + threadIdx.x;
  if (e >= m * 3) return;
  int i = e / 3, c = e % 3;
  int j = fidx[b * m + i];
  out[((size_t)b * m + i) * 3 + c] = xyz[((size_t)b * N + j) * 3 + c];
}

__global__ void rel_pos_kernel(const float* __restrict__ xyz,
                               const int* __restrict__ idx,
                               float* __restrict__ rel, int N, int K)
{
  int b = blockIdx.y;
  long long e = (long long)blockIdx.x * blockDim.x + threadIdx.x;
  long long total = (long long)N * K * 3;
  if (e >= total) return;
  int c = (int)(e % 3);
  long long nk = e / 3;
  int k = (int)(nk % K);
  int n = (int)(nk / K);
  int j = idx[((size_t)b * N + n) * K + k];
  rel[(((size_t)b * N + n) * K + k) * 3 + c] =
      xyz[((size_t)b * N + n) * 3 + c] - xyz[((size_t)b * N + j) * 3 + c];
}

__global__ void qkp_kernel(const float* __restrict__ q,
                           const float* __restrict__ kg,
                           const float* __restrict__ pos,
                           float* __restrict__ t, int N, int K, int C)
{
  int b = blockIdx.y;
  long long e = (long long)blockIdx.x * blockDim.x + threadIdx.x;
  long long total = (long long)N * K * C;
  if (e >= total) return;
  int c = (int)(e % C);
  long long nk = e / C;
  int n = (int)(nk / K);
  size_t o = ((size_t)b * N * K + nk) * C + c;
  t[o] = q[((size_t)b * N + n) * C + c] - kg[o] + pos[o];
}

// out[b,n,c] = sum_k softmax_k(a[b,n,k,c]) * (vg[b,n,k,c] + pos[b,n,k,c])
// K templated so the K logits live in registers: one pass over memory.
template<int KN>
__global__ void attn_agg_kernel(const float* __restrict__ a,
                                const float* __restrict__ vg,
                                const float* __restrict__ pos,
                                float* __restrict__ out, int N, int C)
{
  int b = blockIdx.y;
  long long e = (long long)blockIdx.x * blockDim.x + threadIdx.x;
  if (e >= (long long)N * C) return;
  int c = (int)(e % C), n = (int)(e / C);
  size_t base = (((size_t)b * N + n) * (size_t)KN) * C + c;
  float av[KN];
  #pragma unroll
  for (int k = 0; k < KN; ++k) av[k] = a[base + (size_t)k * C];
  float mx = -3.0e38f;
  #pragma unroll
  for (int k = 0; k < KN; ++k) mx = fmaxf(mx, av[k]);
  float s = 0.0f;
  #pragma unroll
  for (int k = 0; k < KN; ++k) { av[k] = __expf(av[k] - mx); s += av[k]; }
  float inv = 1.0f / s;
  float acc = 0.0f;
  #pragma unroll
  for (int k = 0; k < KN; ++k) {
    size_t o = base + (size_t)k * C;
    acc += av[k] * inv * (vg[o] + pos[o]);
  }
  out[((size_t)b * N + n) * C + c] = acc;
}

__global__ void maxpool_kernel(const float* __restrict__ h,
                               float* __restrict__ out, int M, int K, int C)
{
  long long e = (long long)blockIdx.x * blockDim.x + threadIdx.x;
  if (e >= (long long)M * C) return;
  int c = (int)(e % C), i = (int)(e / C);
  float m = -3.0e38f;
  for (int k = 0; k < K; ++k)
    m = fmaxf(m, h[((size_t)i * K + k) * C + c]);
  out[(size_t)i * C + c] = m;
}

__global__ void interp_kernel(const float* __restrict__ f1,
                              const int* __restrict__ idx,
                              const float* __restrict__ dd,
                              float* __restrict__ out, int Nf, int Nc, int C)
{
  int b = blockIdx.y;
  long long e = (long long)blockIdx.x * blockDim.x + threadIdx.x;
  if (e >= (long long)Nf * C) return;
  int c = (int)(e % C), n = (int)(e / C);
  const int* ip = idx + ((size_t)b * Nf + n) * 3;
  const float* dp = dd + ((size_t)b * Nf + n) * 3;
  float w0 = 1.0f / fmaxf(dp[0], 1e-10f);
  float w1 = 1.0f / fmaxf(dp[1], 1e-10f);
  float w2 = 1.0f / fmaxf(dp[2], 1e-10f);
  float ws = w0 + w1 + w2;
  const float* fb = f1 + (size_t)b * Nc * C;
  out[((size_t)b * Nf + n) * C + c] =
      (w0 * fb[(size_t)ip[0] * C + c] +
       w1 * fb[(size_t)ip[1] * C + c] +
       w2 * fb[(size_t)ip[2] * C + c]) / ws;
}

// =====================================================================
// Host-side orchestration
// =====================================================================
struct Lin { const float* b; const float* w; int din, dout; };
struct PTB { Lin att1, att2, k, lin1, lin2, pos1, pos2, q, v; };

struct Arena {
  char* base; size_t off;
  float* f(size_t n) { float* r = (float*)(base + off); off += ((n * 4 + 255) / 256) * 256; return r; }
  int*  i32(size_t n) { int* r = (int*)(base + off); off += ((n * 4 + 255) / 256) * 256; return r; }
};

static void gemm(hipStream_t st, const float* X, const Lin& L, float* Y,
                 int M, bool relu, const float* res)
{
  dim3 g((M + 15) / 16, (L.dout + 15) / 16), b(32);
  bool fast = (M % 16 == 0) && (L.din % 32 == 0) && (L.dout % 16 == 0);
  if (fast) {
    if (relu) {
      if (res) gemm_wmma_fast<true, true ><<<g, b, 0, st>>>(X, L.w, L.b, res, Y, M, L.din, L.dout);
      else     gemm_wmma_fast<true, false><<<g, b, 0, st>>>(X, L.w, L.b, nullptr, Y, M, L.din, L.dout);
    } else {
      if (res) gemm_wmma_fast<false, true ><<<g, b, 0, st>>>(X, L.w, L.b, res, Y, M, L.din, L.dout);
      else     gemm_wmma_fast<false, false><<<g, b, 0, st>>>(X, L.w, L.b, nullptr, Y, M, L.din, L.dout);
    }
  } else {
    if (relu) {
      if (res) gemm_wmma_edge<true, true ><<<g, b, 0, st>>>(X, L.w, L.b, res, Y, M, L.din, L.dout);
      else     gemm_wmma_edge<true, false><<<g, b, 0, st>>>(X, L.w, L.b, nullptr, Y, M, L.din, L.dout);
    } else {
      if (res) gemm_wmma_edge<false, true ><<<g, b, 0, st>>>(X, L.w, L.b, res, Y, M, L.din, L.dout);
      else     gemm_wmma_edge<false, false><<<g, b, 0, st>>>(X, L.w, L.b, nullptr, Y, M, L.din, L.dout);
    }
  }
}

static void knn_launch(hipStream_t st, const float* q, const float* r,
                       int Nq, int Nr, int K, int* idx, float* dst)
{
  dim3 g((Nq + 127) / 128, BB), b(128);
  switch (K) {
    case 16: knn_kernel<16, false><<<g, b, 0, st>>>(q, r, Nq, Nr, idx, nullptr); break;
    case 8:  knn_kernel<8,  false><<<g, b, 0, st>>>(q, r, Nq, Nr, idx, nullptr); break;
    case 4:  knn_kernel<4,  false><<<g, b, 0, st>>>(q, r, Nq, Nr, idx, nullptr); break;
    case 3:  knn_kernel<3,  true ><<<g, b, 0, st>>>(q, r, Nq, Nr, idx, dst);     break;
    case 2:  knn_kernel<2,  false><<<g, b, 0, st>>>(q, r, Nq, Nr, idx, nullptr); break;
    default: break;
  }
}

static inline dim3 g1(long long total, int tpb, int gy)
{ return dim3((unsigned)((total + tpb - 1) / tpb), (unsigned)gy); }

static void attn_launch(hipStream_t st, const float* a, const float* vg,
                        const float* pos, float* out, int N, int K, int C)
{
  dim3 g = g1((long long)N * C, 256, BB);
  switch (K) {
    case 16: attn_agg_kernel<16><<<g, 256, 0, st>>>(a, vg, pos, out, N, C); break;
    case 8:  attn_agg_kernel<8 ><<<g, 256, 0, st>>>(a, vg, pos, out, N, C); break;
    case 4:  attn_agg_kernel<4 ><<<g, 256, 0, st>>>(a, vg, pos, out, N, C); break;
    case 2:  attn_agg_kernel<2 ><<<g, 256, 0, st>>>(a, vg, pos, out, N, C); break;
    default: break;
  }
}

static void run_ptb(hipStream_t st, const PTB& p, const float* xyz,
                    const float* fin, float* fout, int N, int K, int d, Arena ar)
{
  const int M = BB * N;
  const long long MK = (long long)M * K;
  float* x   = ar.f((size_t)M * d);
  float* qb  = ar.f((size_t)M * d);
  float* kb  = ar.f((size_t)M * d);
  float* vb  = ar.f((size_t)M * d);
  float* sb  = ar.f((size_t)M * d);
  int*   idx = ar.i32((size_t)MK);
  float* rel = ar.f((size_t)MK * 3);
  float* kg  = ar.f((size_t)MK * d);
  float* vg  = ar.f((size_t)MK * d);
  float* pos = ar.f((size_t)MK * d);
  float* t1  = ar.f((size_t)MK * d);  // pos1-out -> t -> a  (reused)
  float* t2  = ar.f((size_t)MK * d);  // relu(att1) out

  gemm(st, fin, p.lin1, x, M, false, nullptr);
  knn_launch(st, xyz, xyz, N, N, K, idx, nullptr);
  gemm(st, x, p.q, qb, M, false, nullptr);
  gemm(st, x, p.k, kb, M, false, nullptr);
  gemm(st, x, p.v, vb, M, false, nullptr);
  gather_feat_kernel<<<g1((long long)N * K * d, 256, BB), 256, 0, st>>>(kb, idx, kg, N, N, K, d);
  gather_feat_kernel<<<g1((long long)N * K * d, 256, BB), 256, 0, st>>>(vb, idx, vg, N, N, K, d);
  rel_pos_kernel<<<g1((long long)N * K * 3, 256, BB), 256, 0, st>>>(xyz, idx, rel, N, K);
  gemm(st, rel, p.pos1, t1, (int)MK, true, nullptr);     // relu(pos1(rel))
  gemm(st, t1, p.pos2, pos, (int)MK, false, nullptr);    // pos
  qkp_kernel<<<g1((long long)N * K * d, 256, BB), 256, 0, st>>>(qb, kg, pos, t1, N, K, d);
  gemm(st, t1, p.att1, t2, (int)MK, true, nullptr);      // relu(att1(t))
  gemm(st, t2, p.att2, t1, (int)MK, false, nullptr);     // a
  attn_launch(st, t1, vg, pos, sb, N, K, d);
  gemm(st, sb, p.lin2, fout, M, false, fin);             // feat + lin2(sum)
}

static void run_tdb(hipStream_t st, const Lin& mlp, const float* xyz,
                    const float* fin, float* nxyz, float* nfeat,
                    int N, int m, int K, Arena ar)
{
  int* fidx = ar.i32((size_t)BB * m);
  fps_kernel<<<BB, 256, 0, st>>>(xyz, N, m, fidx);
  gather_xyz_kernel<<<g1((long long)m * 3, 256, BB), 256, 0, st>>>(xyz, fidx, nxyz, N, m);
  int* idx = ar.i32((size_t)BB * m * K);
  knn_launch(st, nxyz, xyz, m, N, K, idx, nullptr);
  float* fg = ar.f((size_t)BB * m * K * mlp.din);
  gather_feat_kernel<<<g1((long long)m * K * mlp.din, 256, BB), 256, 0, st>>>(fin, idx, fg, m, N, K, mlp.din);
  float* h = ar.f((size_t)BB * m * K * mlp.dout);
  gemm(st, fg, mlp, h, BB * m * K, true, nullptr);
  maxpool_kernel<<<g1((long long)BB * m * mlp.dout, 256, 1), 256, 0, st>>>(h, nfeat, BB * m, K, mlp.dout);
}

static void run_tub(hipStream_t st, const Lin& l1, const Lin& l2,
                    const float* xyz_c, const float* f_c, int Nc,
                    const float* sxyz, const float* sfeat, int Nf,
                    float* fout, Arena ar)
{
  float* f1 = ar.f((size_t)BB * Nc * l1.dout);
  gemm(st, f_c, l1, f1, BB * Nc, false, nullptr);
  int*   idx = ar.i32((size_t)BB * Nf * 3);
  float* dd  = ar.f((size_t)BB * Nf * 3);
  knn_launch(st, sxyz, xyz_c, Nf, Nc, 3, idx, dd);
  float* interp = ar.f((size_t)BB * Nf * l1.dout);
  interp_kernel<<<g1((long long)Nf * l1.dout, 256, BB), 256, 0, st>>>(f1, idx, dd, interp, Nf, Nc, l1.dout);
  gemm(st, sfeat, l2, fout, BB * Nf, false, interp);     // interp + lin2(skip)
}

extern "C" void kernel_launch(void* const* d_in, const int* in_sizes, int n_in,
                              void* d_out, int out_size, void* d_ws, size_t ws_size,
                              hipStream_t stream)
{
  (void)in_sizes; (void)n_in; (void)out_size; (void)ws_size;
  const float* points = (const float*)d_in[0];

  // ---- bind params: JAX pytree leaf order = sorted dict keys, 'b' < 'w' ----
  int ii = 1;
  auto take = [&](int din, int dout) {
    Lin L; L.b = (const float*)d_in[ii++]; L.w = (const float*)d_in[ii++];
    L.din = din; L.dout = dout; return L;
  };
  auto takePTB = [&](int d) {
    PTB p;
    p.att1 = take(d, d); p.att2 = take(d, d); p.k = take(d, d);
    p.lin1 = take(d, d); p.lin2 = take(d, d);
    p.pos1 = take(3, d); p.pos2 = take(d, d);
    p.q = take(d, d); p.v = take(d, d);
    return p;
  };
  Lin lin    = take(6, 32);
  Lin lin2g  = take(512, 512);
  Lin mlpH   = take(32, 13);
  PTB ptb0 = takePTB(32),  ptb1 = takePTB(64),  ptb2 = takePTB(128);
  PTB ptb3 = takePTB(256), ptb4 = takePTB(512), ptb5 = takePTB(512);
  PTB ptb6 = takePTB(256), ptb7 = takePTB(128), ptb8 = takePTB(64);
  PTB ptb9 = takePTB(32);
  Lin tdb1 = take(32, 64),   tdb2 = take(64, 128);
  Lin tdb3 = take(128, 256), tdb4 = take(256, 512);
  Lin tub6a = take(512, 256), tub6b = take(256, 256);
  Lin tub7a = take(256, 128), tub7b = take(128, 128);
  Lin tub8a = take(128, 64),  tub8b = take(64, 64);
  Lin tub9a = take(64, 32),   tub9b = take(32, 32);

  Arena per{ (char*)d_ws, 0 };
  Arena scr{ (char*)d_ws, 16ull << 20 };   // scratch starts at +16MB

  float* xyz0   = (float*)d_out;                       // output 0: B*4096*3
  float* logits = (float*)d_out + (size_t)BB * NPTS0 * 3;

  extract_xyz_kernel<<<g1((long long)NPTS0 * 3, 256, BB), 256, 0, stream>>>(points, xyz0, NPTS0);

  // encoder
  float* featA = per.f((size_t)BB * 4096 * 32);
  gemm(stream, points, lin, featA, BB * 4096, false, nullptr);
  float* s0f = per.f((size_t)BB * 4096 * 32);
  run_ptb(stream, ptb0, xyz0, featA, s0f, 4096, 16, 32, scr);

  float* xyz1 = per.f((size_t)BB * 1024 * 3);
  float* f1a  = per.f((size_t)BB * 1024 * 64);
  run_tdb(stream, tdb1, xyz0, s0f, xyz1, f1a, 4096, 1024, 16, scr);
  float* s1f = per.f((size_t)BB * 1024 * 64);
  run_ptb(stream, ptb1, xyz1, f1a, s1f, 1024, 16, 64, scr);

  float* xyz2 = per.f((size_t)BB * 256 * 3);
  float* f2a  = per.f((size_t)BB * 256 * 128);
  run_tdb(stream, tdb2, xyz1, s1f, xyz2, f2a, 1024, 256, 8, scr);
  float* s2f = per.f((size_t)BB * 256 * 128);
  run_ptb(stream, ptb2, xyz2, f2a, s2f, 256, 8, 128, scr);

  float* xyz3 = per.f((size_t)BB * 64 * 3);
  float* f3a  = per.f((size_t)BB * 64 * 256);
  run_tdb(stream, tdb3, xyz2, s2f, xyz3, f3a, 256, 64, 4, scr);
  float* s3f = per.f((size_t)BB * 64 * 256);
  run_ptb(stream, ptb3, xyz3, f3a, s3f, 64, 4, 256, scr);

  float* xyz4 = per.f((size_t)BB * 16 * 3);
  float* f4a  = per.f((size_t)BB * 16 * 512);
  run_tdb(stream, tdb4, xyz3, s3f, xyz4, f4a, 64, 16, 2, scr);
  float* f4b = per.f((size_t)BB * 16 * 512);
  run_ptb(stream, ptb4, xyz4, f4a, f4b, 16, 2, 512, scr);

  // bottleneck
  float* f4c = per.f((size_t)BB * 16 * 512);
  gemm(stream, f4b, lin2g, f4c, BB * 16, false, nullptr);
  float* f4d = per.f((size_t)BB * 16 * 512);
  run_ptb(stream, ptb5, xyz4, f4c, f4d, 16, 2, 512, scr);

  // decoder
  float* u3 = per.f((size_t)BB * 64 * 256);
  run_tub(stream, tub6a, tub6b, xyz4, f4d, 16, xyz3, s3f, 64, u3, scr);
  float* u3b = per.f((size_t)BB * 64 * 256);
  run_ptb(stream, ptb6, xyz3, u3, u3b, 64, 4, 256, scr);

  float* u2 = per.f((size_t)BB * 256 * 128);
  run_tub(stream, tub7a, tub7b, xyz3, u3b, 64, xyz2, s2f, 256, u2, scr);
  float* u2b = per.f((size_t)BB * 256 * 128);
  run_ptb(stream, ptb7, xyz2, u2, u2b, 256, 8, 128, scr);

  float* u1 = per.f((size_t)BB * 1024 * 64);
  run_tub(stream, tub8a, tub8b, xyz2, u2b, 256, xyz1, s1f, 1024, u1, scr);
  float* u1b = per.f((size_t)BB * 1024 * 64);
  run_ptb(stream, ptb8, xyz1, u1, u1b, 1024, 16, 64, scr);

  float* u0 = per.f((size_t)BB * 4096 * 32);
  run_tub(stream, tub9a, tub9b, xyz1, u1b, 1024, xyz0, s0f, 4096, u0, scr);
  float* u0b = per.f((size_t)BB * 4096 * 32);
  run_ptb(stream, ptb9, xyz0, u0, u0b, 4096, 16, 32, scr);

  // head
  gemm(stream, u0b, mlpH, logits, BB * 4096, false, nullptr);
}
